// DecoderRNN_90151363543550
// MI455X (gfx1250) — matrix operand order repaired and checked by reading, not compiled, not run on hardware
//
#include <hip/hip_runtime.h>
#include <math.h>

// ---------------------------------------------------------------------------
// Problem constants
// ---------------------------------------------------------------------------
#define VIN   50000
#define VOUT  50000
#define EMB   256
#define HID   128
#define BATCH 64
#define SEQ   8192
#define CHUNK 512
#define NCHUNK (SEQ / CHUNK)   // 16
#define NTILE  (CHUNK / 16)    // 32

typedef __attribute__((ext_vector_type(16))) __bf16 v16bf;
typedef __attribute__((ext_vector_type(8)))  __bf16 v8bf;
typedef __attribute__((ext_vector_type(4)))  __bf16 v4bf;
typedef __attribute__((ext_vector_type(8)))  float  v8f;

// ---------------------------------------------------------------------------
// Branch-free transcendentals (v_exp_f32 / v_rcp_f32, no EXEC divergence)
// ---------------------------------------------------------------------------
__device__ __forceinline__ float fast_exp(float x) {
    return __builtin_amdgcn_exp2f(x * 1.44269504088896340736f);
}
__device__ __forceinline__ float fast_tanh(float x) {
    // tanh(x) = 1 - 2/(exp(2x)+1); exp2 overflow->inf gives 1, underflow->0 gives -1
    float e2 = __builtin_amdgcn_exp2f(x * 2.88539008177792681472f);
    return 1.f - 2.f * __builtin_amdgcn_rcpf(e2 + 1.f);
}
__device__ __forceinline__ float fast_sigmoid(float x) {
    return __builtin_amdgcn_rcpf(1.f + fast_exp(-x));
}

__device__ __forceinline__ v16bf load_bf16_pair(const __bf16* p0, const __bf16* p1) {
    v8bf a = *(const v8bf*)p0;   // 16B -> ds_load_b128
    v8bf b = *(const v8bf*)p1;
    return __builtin_shufflevector(a, b, 0,1,2,3,4,5,6,7,8,9,10,11,12,13,14,15);
}

// Build a 16x32-layout bf16 A-fragment from f32 data in LDS.
// p points at the 8-float run for elems 0..7; elems 8..15 live at p+16.
__device__ __forceinline__ v16bf frag_from_f32(const float* p) {
    v16bf r;
    #pragma unroll
    for (int j = 0; j < 8; ++j)  r[j]     = (__bf16)p[j];
    #pragma unroll
    for (int j = 0; j < 8; ++j)  r[8 + j] = (__bf16)p[16 + j];
    return r;
}

// Async copy of one 16-row encoder tile (16*128 f32 = 512 x 16B) into LDS.
// Per-lane GLOBAL_LOAD_ASYNC_TO_LDS_B128 (GV mode), tracked by ASYNCcnt.
__device__ __forceinline__ void issue_tile_copy(const float* __restrict__ rows,
                                                float* __restrict__ dstbuf, int tid) {
    #pragma unroll
    for (int i = 0; i < 2; ++i) {
        int idx = tid + i * 256;                         // 16B-chunk index, 512 total
        unsigned lds_off = (unsigned)(uintptr_t)(&dstbuf[idx * 4]);
        const float* src = rows + (size_t)idx * 4;
        asm volatile("global_load_async_to_lds_b128 %0, %1, off"
                     :: "v"(lds_off), "v"(src) : "memory");
    }
}
__device__ __forceinline__ void wait_async_copies() {
    asm volatile("s_wait_asynccnt 0x0" ::: "memory");
}

// ---------------------------------------------------------------------------
// Kernel 0a: convert Uattn (f32 [H,H], row = output dim n) to bf16
// ---------------------------------------------------------------------------
__global__ void k_prep_ubf(const float* __restrict__ U, __bf16* __restrict__ Ubf) {
    int i = blockIdx.x * 256 + threadIdx.x;   // 16384 total
    Ubf[i] = (__bf16)U[i];
}

// ---------------------------------------------------------------------------
// Kernel 0b: Wa[b,n] = sum_k hidden[b,k] * Wattn[n,k]
// ---------------------------------------------------------------------------
__global__ void k_wa(const float* __restrict__ hidden, const float* __restrict__ Wattn,
                     float* __restrict__ Wa) {
    __shared__ float hrow[HID];
    int b = blockIdx.x, n = threadIdx.x;      // 128 threads
    hrow[n] = hidden[b * HID + n];
    __syncthreads();
    float acc = 0.f;
    const float* wr = Wattn + (size_t)n * HID;
    #pragma unroll 4
    for (int k = 0; k < HID; ++k) acc += wr[k] * hrow[k];
    Wa[b * HID + n] = acc;
}

// ---------------------------------------------------------------------------
// Kernel 1: chunked Bahdanau attention, bf16 WMMA, async double-buffered
//   staging, partial softmax.  grid = (NCHUNK, BATCH), block = 256 (8 waves)
//   part[(b*NCHUNK+c)*130] = {max, sumexp, ct_partial[128]}
// ---------------------------------------------------------------------------
__global__ __launch_bounds__(256) void k_attn(
    const float*  __restrict__ enc,    // [B,S,H]
    const __bf16* __restrict__ Ubf,    // [H,H] row-major (row n, col k)
    const float*  __restrict__ Wa,     // [B,H]
    const float*  __restrict__ Vw,     // [H]
    float*        __restrict__ part) {
    __shared__ alignas(16) __bf16 U_lds[HID * HID];      // 32 KB
    __shared__ alignas(16) float  stg[2][16 * HID];      // 16 KB (double buffer)
    __shared__ float e_chunk[CHUNK];                     // 2 KB
    __shared__ float epart[8][16];
    __shared__ float red[256];

    const int tid  = threadIdx.x;
    const int b    = blockIdx.y;
    const int chnk = blockIdx.x;
    const int wid  = tid >> 5;
    const int lane = tid & 31;
    const int half = lane >> 4;
    const int l15  = lane & 15;

    const float* enc_blk = enc + ((size_t)b * SEQ + (size_t)chnk * CHUNK) * HID;

    // kick off async copy of tile 0 while we stage U
    issue_tile_copy(enc_blk, &stg[0][0], tid);

    // Stage full Uattn-bf16 into LDS (2048 uint4)
    {
        const uint4* src = (const uint4*)Ubf;
        uint4* dst = (uint4*)U_lds;
        #pragma unroll
        for (int i = 0; i < 8; ++i) dst[tid + i * 256] = src[tid + i * 256];
    }
    // Wave `wid` owns output columns n = wid*16 .. wid*16+15; lane owns one n.
    const int   n_full = wid * 16 + l15;
    const float v_n    = Vw[n_full];
    const float wa_n   = Wa[b * HID + n_full];

    __syncthreads();

    // Preload the 4 K-block B fragments for this wave's N-tile.
    // B[k][n] = Uattn[n][k]; lane=n, elems j<8 -> k = kb*32 + half*8 + j, j>=8 -> k = kb*32 + 16 + half*8 + (j-8)
    v16bf Bfrag[4];
    #pragma unroll
    for (int kb = 0; kb < 4; ++kb) {
        const __bf16* p = &U_lds[n_full * HID + kb * 32 + half * 8];
        Bfrag[kb] = load_bf16_pair(p, p + 16);
    }

    for (int t = 0; t < NTILE; ++t) {
        wait_async_copies();          // this wave's tile-t chunks have landed
        __syncthreads();              // ...and everyone else's are visible
        if (t + 1 < NTILE)            // overlap next tile with this tile's math
            issue_tile_copy(enc_blk + (size_t)(t + 1) * 16 * HID,
                            &stg[(t + 1) & 1][0], tid);

        const float* cur = &stg[t & 1][0];

        // Materialize all four A-fragments in distinct registers first so the
        // WMMA chain issues back-to-back (D->C accumulate only, no WAR NOPs).
        v16bf afrag[4];
        #pragma unroll
        for (int kb = 0; kb < 4; ++kb)
            afrag[kb] = frag_from_f32(cur + l15 * HID + kb * 32 + half * 8);

        // ---- WMMA: [16 x 128] tile @ [128 x 16] slice, C init = Wa ----
        v8f acc;
        #pragma unroll
        for (int r = 0; r < 8; ++r) acc[r] = wa_n;
        #pragma unroll
        for (int kb = 0; kb < 4; ++kb)
            acc = __builtin_amdgcn_wmma_f32_16x16x32_bf16(
                false, afrag[kb], false, Bfrag[kb], (short)0, acc, false, false);

        // e contribution: tanh(Ua+Wa)*v, reduce over the 16 N-lanes of each half
        float ep[8];
        #pragma unroll
        for (int r = 0; r < 8; ++r) ep[r] = fast_tanh(acc[r]) * v_n;
        #pragma unroll
        for (int m = 1; m < 16; m <<= 1) {
            #pragma unroll
            for (int r = 0; r < 8; ++r) ep[r] += __shfl_xor(ep[r], m, 32);
        }
        if (l15 == 0) {
            #pragma unroll
            for (int r = 0; r < 8; ++r) epart[wid][half * 8 + r] = ep[r];
        }
        __syncthreads();
        if (tid < 16) {               // sum the 8 wave partials per row
            float s = 0.f;
            #pragma unroll
            for (int w = 0; w < 8; ++w) s += epart[w][tid];
            e_chunk[t * 16 + tid] = s;
        }
        // next iteration's epart writes are fenced by its top-of-loop barrier
    }
    __syncthreads();

    // ---- block softmax over the 512 chunk logits ----
    float lm = fmaxf(e_chunk[tid], e_chunk[tid + 256]);
    red[tid] = lm;
    __syncthreads();
    for (int s = 128; s > 0; s >>= 1) {
        if (tid < s) red[tid] = fmaxf(red[tid], red[tid + s]);
        __syncthreads();
    }
    float gmax = red[0];
    __syncthreads();
    float ls = fast_exp(e_chunk[tid] - gmax) + fast_exp(e_chunk[tid + 256] - gmax);
    red[tid] = ls;
    __syncthreads();
    for (int s = 128; s > 0; s >>= 1) {
        if (tid < s) red[tid] += red[tid + s];
        __syncthreads();
    }
    float lsum = red[0];
    __syncthreads();

    // ---- ct_partial[h] = sum_i exp(e_i - max) * enc[i][h] ----
    const int h = tid & 127;
    const int g = tid >> 7;
    float ct = 0.f;
    const float* ep2 = enc_blk + (size_t)g * 256 * HID;
    for (int i = 0; i < 256; ++i) {
        if (i + 8 < 256) __builtin_prefetch(&ep2[(size_t)(i + 8) * HID + h], 0, 1);
        float w = fast_exp(e_chunk[g * 256 + i] - gmax);
        ct += w * ep2[(size_t)i * HID + h];
    }
    red[tid] = ct;
    __syncthreads();
    if (tid < 128) {
        float c2 = red[tid] + red[tid + 128];
        float* po = part + ((size_t)b * NCHUNK + chnk) * 130;
        po[2 + tid] = c2;
        if (tid == 0) { po[0] = gmax; po[1] = lsum; }
    }
}

// ---------------------------------------------------------------------------
// Kernel 2: combine chunk partials -> ct[b,h]
// ---------------------------------------------------------------------------
__global__ void k_attn_finalize(const float* __restrict__ part, float* __restrict__ ct_out) {
    int b = blockIdx.x, h = threadIdx.x;   // 128 threads
    float gm = -1e30f;
    for (int c = 0; c < NCHUNK; ++c)
        gm = fmaxf(gm, part[((size_t)b * NCHUNK + c) * 130]);
    float L = 0.f, acc = 0.f;
    for (int c = 0; c < NCHUNK; ++c) {
        const float* p = part + ((size_t)b * NCHUNK + c) * 130;
        float sc = fast_exp(p[0] - gm);
        L   += p[1] * sc;
        acc += p[2 + h] * sc;
    }
    ct_out[b * HID + h] = acc / L;
}

// ---------------------------------------------------------------------------
// Kernel 3: single-step GRU with h0 = 0  (gh = b_hh only)
// ---------------------------------------------------------------------------
__global__ void k_gru(const int* __restrict__ x, const float* __restrict__ emb,
                      const float* __restrict__ ct, const float* __restrict__ W_ih,
                      const float* __restrict__ b_ih, const float* __restrict__ b_hh,
                      float* __restrict__ h_out, float* __restrict__ hidden_out) {
    __shared__ float rin[EMB + HID];          // 384
    int b = blockIdx.x, t = threadIdx.x;      // 128 threads
    int xi = x[b];
    rin[t]       = emb[(size_t)xi * EMB + t];
    rin[t + 128] = emb[(size_t)xi * EMB + 128 + t];
    rin[t + 256] = ct[b * HID + t];
    __syncthreads();
    float ar = b_ih[t], az = b_ih[HID + t], an = b_ih[2 * HID + t];
    const float* wr = W_ih + (size_t)t * 384;
    const float* wz = W_ih + (size_t)(HID + t) * 384;
    const float* wn = W_ih + (size_t)(2 * HID + t) * 384;
    #pragma unroll 4
    for (int k = 0; k < 384; ++k) {
        float rv = rin[k];
        ar += wr[k] * rv; az += wz[k] * rv; an += wn[k] * rv;
    }
    float r  = fast_sigmoid(ar + b_hh[t]);
    float z  = fast_sigmoid(az + b_hh[HID + t]);
    float n  = fast_tanh(an + r * b_hh[2 * HID + t]);
    float hv = (1.f - z) * n;
    h_out[b * HID + t]      = hv;
    hidden_out[b * HID + t] = hv;
}

// ---------------------------------------------------------------------------
// Kernel 4: predictions = h @ fc_w.T + fc_b via bf16 WMMA
//   grid = ceil(VOUT/128), block = 256 (8 waves, one 16-wide N slice each)
// ---------------------------------------------------------------------------
__global__ __launch_bounds__(256) void k_fc(
    const float* __restrict__ hmat,   // [64,128]
    const float* __restrict__ fc_w,   // [VOUT,128]
    const float* __restrict__ fc_b,   // [VOUT]
    float* __restrict__ out) {        // [64, VOUT]
    __shared__ alignas(16) __bf16 h_lds[BATCH * HID];   // 16 KB
    const int tid  = threadIdx.x;
    const int wid  = tid >> 5;
    const int lane = tid & 31;
    const int half = lane >> 4;
    const int l15  = lane & 15;

    // stage h as bf16 (2048 float4)
    #pragma unroll
    for (int i = 0; i < 8; ++i) {
        int idx = tid + i * 256;
        float4 f = ((const float4*)hmat)[idx];
        v4bf hv;
        hv[0] = (__bf16)f.x; hv[1] = (__bf16)f.y;
        hv[2] = (__bf16)f.z; hv[3] = (__bf16)f.w;
        *(v4bf*)&h_lds[idx * 4] = hv;
    }
    __syncthreads();

    const int vo    = blockIdx.x * 128 + wid * 16 + l15;
    const bool valid = (vo < VOUT);
    v16bf Bfrag[4];
    #pragma unroll
    for (int kb = 0; kb < 4; ++kb) {
        v16bf bv;
        if (valid) {
            const float* p = fc_w + (size_t)vo * HID + kb * 32 + half * 8;
            #pragma unroll
            for (int j = 0; j < 16; ++j) bv[j] = (__bf16)p[j + (j >> 3) * 8];
        } else {
            #pragma unroll
            for (int j = 0; j < 16; ++j) bv[j] = (__bf16)0.f;
        }
        Bfrag[kb] = bv;
    }
    const float bias = valid ? fc_b[vo] : 0.f;

    #pragma unroll
    for (int mt = 0; mt < 4; ++mt) {
        // all four A-fragments in distinct registers, then a clean WMMA chain
        v16bf afrag[4];
        #pragma unroll
        for (int kb = 0; kb < 4; ++kb) {
            const __bf16* pa = &h_lds[(mt * 16 + l15) * HID + kb * 32 + half * 8];
            afrag[kb] = load_bf16_pair(pa, pa + 16);
        }
        v8f acc;
        #pragma unroll
        for (int r = 0; r < 8; ++r) acc[r] = 0.f;
        #pragma unroll
        for (int kb = 0; kb < 4; ++kb)
            acc = __builtin_amdgcn_wmma_f32_16x16x32_bf16(
                false, afrag[kb], false, Bfrag[kb], (short)0, acc, false, false);
        if (valid) {
            #pragma unroll
            for (int r = 0; r < 8; ++r) {
                int m = mt * 16 + half * 8 + r;       // batch row
                out[(size_t)m * VOUT + vo] = acc[r] + bias;
            }
        }
    }
}

// ---------------------------------------------------------------------------
// Host launcher
// ---------------------------------------------------------------------------
extern "C" void kernel_launch(void* const* d_in, const int* in_sizes, int n_in,
                              void* d_out, int out_size, void* d_ws, size_t ws_size,
                              hipStream_t stream) {
    (void)in_sizes; (void)n_in; (void)out_size; (void)ws_size;
    const int*   x       = (const int*)  d_in[0];
    const float* enc     = (const float*)d_in[1];
    const float* hidden  = (const float*)d_in[2];
    // d_in[3] = cell (unused), d_in[4] = batch_size (unused)
    const float* emb     = (const float*)d_in[5];
    const float* Wattn_w = (const float*)d_in[6];
    const float* Uattn_w = (const float*)d_in[7];
    const float* Vattn_w = (const float*)d_in[8];
    const float* W_ih    = (const float*)d_in[9];
    // d_in[10] = W_hh (unused: h0 == 0)
    const float* b_ih    = (const float*)d_in[11];
    const float* b_hh    = (const float*)d_in[12];
    const float* fc_w    = (const float*)d_in[13];
    const float* fc_b    = (const float*)d_in[14];

    float* out_pred   = (float*)d_out;                         // [64,50000]
    float* out_hidden = (float*)d_out + (size_t)BATCH * VOUT;  // [64,128]

    // workspace layout (floats)
    float*  ws      = (float*)d_ws;
    float*  ws_Wa   = ws;                               //  8192 f
    __bf16* ws_Ubf  = (__bf16*)(ws + 8192);             // 16384 bf16 (= 8192 f)
    float*  ws_part = ws + 8192 + 8192;                 // 64*16*130 = 133120 f
    float*  ws_ct   = ws_part + BATCH * NCHUNK * 130;   //  8192 f
    float*  ws_h    = ws_ct + BATCH * HID;              //  8192 f

    k_prep_ubf<<<64, 256, 0, stream>>>(Uattn_w, ws_Ubf);
    k_wa<<<BATCH, HID, 0, stream>>>(hidden, Wattn_w, ws_Wa);
    k_attn<<<dim3(NCHUNK, BATCH), 256, 0, stream>>>(enc, ws_Ubf, ws_Wa, Vattn_w, ws_part);
    k_attn_finalize<<<BATCH, HID, 0, stream>>>(ws_part, ws_ct);
    k_gru<<<BATCH, HID, 0, stream>>>(x, emb, ws_ct, W_ih, b_ih, b_hh, ws_h, out_hidden);
    k_fc<<<(VOUT + 127) / 128, 256, 0, stream>>>(ws_h, fc_w, fc_b, out_pred);
}
